// MeanShift_55997783605366
// MI455X (gfx1250) — matrix operand order repaired and checked by reading, not compiled, not run on hardware
//
#include <hip/hip_runtime.h>
#include <hip/hip_bf16.h>
#include <math.h>

// Mean-shift clustering for MI455X (gfx1250, wave32, WMMA).
// N=4096 points, D=256. Compute-bound: all GEMMs run on
// v_wmma_f32_16x16x32_bf16 (bf16 in, f32 accumulate). W (N x N bf16, 32MB)
// is materialized in workspace between the two GEMMs of each iteration --
// it stays resident in the 192MB L2. Workspace requirement: ~42MB.

#define NPTS 4096
#define DIM  256
#define MSITERS 10
#define SHIFT_TH 1e-5f
#define EPSN 1e-8f

typedef __bf16 v16bf __attribute__((ext_vector_type(16)));
typedef __bf16 v8bf  __attribute__((ext_vector_type(8)));
typedef float  v8f   __attribute__((ext_vector_type(8)));

__device__ __forceinline__ __bf16 f2bf(float f) {
  union { float f; unsigned u; } in; in.f = f;
  unsigned r = (in.u + 0x7FFFu + ((in.u >> 16) & 1u)) >> 16;  // RNE
  union { unsigned short s; __bf16 b; } out; out.s = (unsigned short)r;
  return out.b;
}

__device__ __forceinline__ v8f zero_v8f() {
  v8f z; 
#pragma unroll
  for (int i = 0; i < 8; ++i) z[i] = 0.0f;
  return z;
}

// A fragment: 16x32 bf16, row-major source with leading dim lda.
// Lane l (m = l&15): rows row16+m; K = 0..15 (l<16) or 16..31 -- contiguous 32B.
__device__ __forceinline__ v16bf load_frag_a(const __bf16* __restrict__ base,
                                             int lda, int row16, int kbase, int lane) {
  int m = lane & 15;
  int ksel = (lane < 16) ? 0 : 16;
  const __bf16* p = base + (size_t)(row16 + m) * lda + kbase + ksel;
  return *reinterpret_cast<const v16bf*>(p);
}

// B fragment: 32x16 bf16. Source rows (stride ldb) are the COLUMNS of B.
// Lane l (n = l&15): column col16+n; K = 0..15 (l<16) or 16..31 -- contiguous 32B.
__device__ __forceinline__ v16bf load_frag_b(const __bf16* __restrict__ base,
                                             int ldb, int col16, int kbase, int lane) {
  int n = lane & 15;
  int ksel = (lane < 16) ? 0 : 16;
  const __bf16* p = base + (size_t)(col16 + n) * ldb + kbase + ksel;
  return *reinterpret_cast<const v16bf*>(p);
}

__device__ __forceinline__ v8f wmma_bf16(v16bf a, v16bf b, v8f c) {
  return __builtin_amdgcn_wmma_f32_16x16x32_bf16(false, a, false, b, (short)0, c,
                                                 false, false);
}

// ---------------- prep kernels ----------------

// one block per row; block = 256 threads = one dim each
__global__ __launch_bounds__(256) void k_init(const float* __restrict__ x,
                                              float* __restrict__ cent,
                                              __bf16* __restrict__ xn,
                                              __bf16* __restrict__ xbT,
                                              int* shiftInt, int* done) {
  __shared__ float sred[9];
  int row = blockIdx.x;
  int t = threadIdx.x;
  float v = x[(size_t)row * DIM + t];
  float sq = v * v;
#pragma unroll
  for (int m = 1; m < 32; m <<= 1) sq += __shfl_xor(sq, m, 32);
  if ((t & 31) == 0) sred[t >> 5] = sq;
  __syncthreads();
  if (t == 0) {
    float s = 0.0f;
    for (int w = 0; w < 8; ++w) s += sred[w];
    sred[8] = fmaxf(sqrtf(s), EPSN);
  }
  __syncthreads();
  float nrm = sred[8];
  cent[(size_t)row * DIM + t] = v;
  xn[(size_t)row * DIM + t] = f2bf(v / nrm);
  xbT[(size_t)t * NPTS + row] = f2bf(v);
  if (row == 0 && t == 0) { *shiftInt = 0; *done = 0; }
}

__global__ __launch_bounds__(256) void k_norm_c(const float* __restrict__ cent,
                                                __bf16* __restrict__ cn) {
  __shared__ float sred[9];
  int row = blockIdx.x;
  int t = threadIdx.x;
  float v = cent[(size_t)row * DIM + t];
  float sq = v * v;
#pragma unroll
  for (int m = 1; m < 32; m <<= 1) sq += __shfl_xor(sq, m, 32);
  if ((t & 31) == 0) sred[t >> 5] = sq;
  __syncthreads();
  if (t == 0) {
    float s = 0.0f;
    for (int w = 0; w < 8; ++w) s += sred[w];
    sred[8] = fmaxf(sqrtf(s), EPSN);
  }
  __syncthreads();
  cn[(size_t)row * DIM + t] = f2bf(v / sred[8]);
}

__global__ void k_reset_iter(float* rowsum, int* shiftInt) {
  int i = blockIdx.x * 256 + threadIdx.x;
  rowsum[i] = 0.0f;
  if (i == 0) *shiftInt = 0;
}

// ---------------- GEMM1: W = exp((cn.xn^T - 1)/sigma^2), rowsum ----------------
// block = 8 waves (4 row-groups x 2 col-groups), wave tile 32x64, block 128x128.
__global__ __launch_bounds__(256) void k_gemm1(const __bf16* __restrict__ cn,
                                               const __bf16* __restrict__ xn,
                                               __bf16* __restrict__ Wm,
                                               float* __restrict__ rowsum,
                                               const float* __restrict__ sigma) {
  int lane = threadIdx.x & 31;
  int wv = threadIdx.x >> 5;
  int wr = wv & 3, wc = wv >> 2;
  int rowBase = blockIdx.y * 128 + wr * 32;
  int colBase = blockIdx.x * 128 + wc * 64;
  float g = 1.0f / (sigma[0] * sigma[0]);

  v8f acc[2][4];
#pragma unroll
  for (int rt = 0; rt < 2; ++rt)
#pragma unroll
    for (int ct = 0; ct < 4; ++ct) acc[rt][ct] = zero_v8f();

  for (int k = 0; k < DIM; k += 32) {
    v16bf a[2], b[4];
#pragma unroll
    for (int rt = 0; rt < 2; ++rt) a[rt] = load_frag_a(cn, DIM, rowBase + rt * 16, k, lane);
#pragma unroll
    for (int ct = 0; ct < 4; ++ct) b[ct] = load_frag_b(xn, DIM, colBase + ct * 16, k, lane);
#pragma unroll
    for (int rt = 0; rt < 2; ++rt)
#pragma unroll
      for (int ct = 0; ct < 4; ++ct) acc[rt][ct] = wmma_bf16(a[rt], b[ct], acc[rt][ct]);
  }

#pragma unroll
  for (int rt = 0; rt < 2; ++rt) {
    float rsum[8];
#pragma unroll
    for (int r = 0; r < 8; ++r) rsum[r] = 0.0f;
#pragma unroll
    for (int ct = 0; ct < 4; ++ct) {
#pragma unroll
      for (int r = 0; r < 8; ++r) {
        int row = rowBase + rt * 16 + r + ((lane < 16) ? 0 : 8);
        int col = colBase + ct * 16 + (lane & 15);
        float w = __expf((acc[rt][ct][r] - 1.0f) * g);
        Wm[(size_t)row * NPTS + col] = f2bf(w);
        rsum[r] += w;
      }
    }
#pragma unroll
    for (int r = 0; r < 8; ++r) {
      float v = rsum[r];
#pragma unroll
      for (int m = 1; m < 16; m <<= 1) v += __shfl_xor(v, m, 32);  // sum 16 cols
      if ((lane & 15) == 0) {
        int row = rowBase + rt * 16 + r + ((lane < 16) ? 0 : 8);
        atomicAdd(&rowsum[row], v);
      }
    }
  }
}

// ---------------- GEMM2: newc = (W @ x)/rowsum, shift = max row ||d||^2 ----------
// block = 8 waves (2 row-groups x 4 col-groups), wave tile 32x64, block 64x256.
__global__ __launch_bounds__(256) void k_gemm2(const __bf16* __restrict__ Wm,
                                               const __bf16* __restrict__ xbT,
                                               const float* __restrict__ rowsum,
                                               const float* __restrict__ cent,
                                               float* __restrict__ newc,
                                               int* shiftInt) {
  int lane = threadIdx.x & 31;
  int wv = threadIdx.x >> 5;
  int wr = wv >> 2, wc = wv & 3;
  int rowBase = blockIdx.x * 64 + wr * 32;
  int colBase = wc * 64;

  v8f acc[2][4];
#pragma unroll
  for (int rt = 0; rt < 2; ++rt)
#pragma unroll
    for (int ct = 0; ct < 4; ++ct) acc[rt][ct] = zero_v8f();

  for (int k = 0; k < NPTS; k += 32) {
    v16bf a[2], b[4];
#pragma unroll
    for (int rt = 0; rt < 2; ++rt) a[rt] = load_frag_a(Wm, NPTS, rowBase + rt * 16, k, lane);
#pragma unroll
    for (int ct = 0; ct < 4; ++ct) b[ct] = load_frag_b(xbT, NPTS, colBase + ct * 16, k, lane);
#pragma unroll
    for (int rt = 0; rt < 2; ++rt)
#pragma unroll
      for (int ct = 0; ct < 4; ++ct) acc[rt][ct] = wmma_bf16(a[rt], b[ct], acc[rt][ct]);
  }

#pragma unroll
  for (int rt = 0; rt < 2; ++rt) {
    float ss[8];
#pragma unroll
    for (int r = 0; r < 8; ++r) ss[r] = 0.0f;
#pragma unroll
    for (int ct = 0; ct < 4; ++ct) {
#pragma unroll
      for (int r = 0; r < 8; ++r) {
        int row = rowBase + rt * 16 + r + ((lane < 16) ? 0 : 8);
        int col = colBase + ct * 16 + (lane & 15);
        float rs = rowsum[row];
        float val = acc[rt][ct][r] / rs;
        newc[(size_t)row * DIM + col] = val;
        float d = val - cent[(size_t)row * DIM + col];
        ss[r] += d * d;
      }
    }
#pragma unroll
    for (int r = 0; r < 8; ++r) {
      float v = ss[r];
#pragma unroll
      for (int m = 1; m < 16; m <<= 1) v += __shfl_xor(v, m, 32);
      if ((lane & 15) == 0) atomicMax(shiftInt, __float_as_int(v));  // v >= 0: bits monotone
    }
  }
}

__global__ void k_update(float* cent, const float* __restrict__ newc,
                         const int* shiftInt, const int* done) {
  size_t i = (size_t)blockIdx.x * 256 + threadIdx.x;
  float sh = __int_as_float(*shiftInt);  // max squared shift
  bool below = sh < (SHIFT_TH * SHIFT_TH);
  if (!((*done != 0) || below)) cent[i] = newc[i];
}

__global__ void k_done(const int* shiftInt, int* done) {
  float sh = __int_as_float(*shiftInt);
  if (sh < (SHIFT_TH * SHIFT_TH)) *done = 1;
}

__global__ void k_reset_nms(float* counts, int* keepInt) {
  int i = blockIdx.x * 256 + threadIdx.x;
  counts[i] = 0.0f;
  keepInt[i] = 0;
}

// ---------------- NMS pass 1: per-point closest centroid -> counts ----------------
// block = 8 waves x 16 columns = 128 points; each wave scans all 4096 centroid rows.
__global__ __launch_bounds__(256) void k_closest(const __bf16* __restrict__ cn,
                                                 const __bf16* __restrict__ xn,
                                                 float* __restrict__ counts) {
  int lane = threadIdx.x & 31;
  int wv = threadIdx.x >> 5;
  int colBase = blockIdx.x * 128 + wv * 16;

  v16bf bq[8];
#pragma unroll
  for (int kc = 0; kc < 8; ++kc) bq[kc] = load_frag_b(xn, DIM, colBase, kc * 32, lane);

  float bestS = -1e30f;
  int bestI = 0;
  for (int it = 0; it < NPTS / 16; ++it) {
    v8f acc = zero_v8f();
#pragma unroll
    for (int kc = 0; kc < 8; ++kc) {
      v16bf a = load_frag_a(cn, DIM, it * 16, kc * 32, lane);
      acc = wmma_bf16(a, bq[kc], acc);
    }
#pragma unroll
    for (int r = 0; r < 8; ++r) {
      int row = it * 16 + r + ((lane < 16) ? 0 : 8);
      float s = acc[r];
      if (s > bestS) { bestS = s; bestI = row; }  // rows ascend: ties keep first
    }
  }
  // merge the two lane halves (same column, disjoint row sets); tie -> lower index
  float os = __shfl_xor(bestS, 16, 32);
  int oi = __shfl_xor(bestI, 16, 32);
  if (os > bestS || (os == bestS && oi < bestI)) { bestS = os; bestI = oi; }
  if (lane < 16) atomicAdd(&counts[bestI], 1.0f);
}

// ---------------- NMS pass 2: mapped = argmax_j (close_ij * counts_j); keep -------
// block = 8 waves x 16 rows = 128 centroids; counts cached in LDS.
__global__ __launch_bounds__(256) void k_map(const __bf16* __restrict__ cn,
                                             const float* __restrict__ counts,
                                             int* __restrict__ keepInt,
                                             const float* __restrict__ sigma) {
  __shared__ float sc[NPTS];
  for (int i = threadIdx.x; i < NPTS; i += 256) sc[i] = counts[i];
  __syncthreads();

  int lane = threadIdx.x & 31;
  int wv = threadIdx.x >> 5;
  int rowBase = blockIdx.x * 128 + wv * 16;
  float bw = sigma[0];

  v16bf aq[8];
#pragma unroll
  for (int kc = 0; kc < 8; ++kc) aq[kc] = load_frag_a(cn, DIM, rowBase, kc * 32, lane);

  float runS[8];
  int runC[8];
#pragma unroll
  for (int r = 0; r < 8; ++r) { runS[r] = -1.0f; runC[r] = 0; }

  for (int jt = 0; jt < NPTS / 16; ++jt) {
    v8f acc = zero_v8f();
#pragma unroll
    for (int kc = 0; kc < 8; ++kc) {
      v16bf b = load_frag_b(cn, DIM, jt * 16, kc * 32, lane);
      acc = wmma_bf16(aq[kc], b, acc);
    }
    int colMine = jt * 16 + (lane & 15);
    float cntMine = sc[colMine];
#pragma unroll
    for (int r = 0; r < 8; ++r) {
      float s = acc[r];
      float score = ((2.0f * (1.0f - s)) < bw) ? cntMine : 0.0f;
      int c = colMine;
#pragma unroll
      for (int m = 1; m < 16; m <<= 1) {  // butterfly over the 16 columns
        float osc = __shfl_xor(score, m, 32);
        int oc = __shfl_xor(c, m, 32);
        if (osc > score || (osc == score && oc < c)) { score = osc; c = oc; }
      }
      if (score > runS[r]) { runS[r] = score; runC[r] = c; }  // tiles ascend
    }
  }
  if ((lane & 15) == 0) {
#pragma unroll
    for (int r = 0; r < 8; ++r) {
      int row = rowBase + r + ((lane < 16) ? 0 : 8);
      if (sc[row] > 0.0f) atomicMax(&keepInt[runC[r]], 1);
    }
  }
}

__global__ void k_keep(const int* __restrict__ keepInt, float* __restrict__ out) {
  int i = blockIdx.x * 256 + threadIdx.x;
  out[i] = (keepInt[i] != 0) ? 1.0f : 0.0f;
}

// ---------------- launch ----------------

extern "C" void kernel_launch(void* const* d_in, const int* in_sizes, int n_in,
                              void* d_out, int out_size, void* d_ws, size_t ws_size,
                              hipStream_t stream) {
  (void)in_sizes; (void)n_in; (void)out_size; (void)ws_size;
  const float* x = (const float*)d_in[0];
  const float* sigma = (const float*)d_in[1];

  float* cent = (float*)d_out;                       // N*D centroids
  float* keepOut = (float*)d_out + (size_t)NPTS * DIM;  // N keep mask (0/1)

  char* ws = (char*)d_ws;
  size_t oW = 0;
  size_t oXn = oW + (size_t)NPTS * NPTS * sizeof(__bf16);     // 32 MB
  size_t oXbT = oXn + (size_t)NPTS * DIM * sizeof(__bf16);    // +2 MB
  size_t oCn = oXbT + (size_t)NPTS * DIM * sizeof(__bf16);    // +2 MB
  size_t oNewc = oCn + (size_t)NPTS * DIM * sizeof(__bf16);   // +2 MB
  size_t oRowsum = oNewc + (size_t)NPTS * DIM * sizeof(float);  // +4 MB
  size_t oCounts = oRowsum + (size_t)NPTS * sizeof(float);
  size_t oKeep = oCounts + (size_t)NPTS * sizeof(float);
  size_t oShift = oKeep + (size_t)NPTS * sizeof(int);
  size_t oDone = oShift + 256;

  __bf16* Wm = (__bf16*)(ws + oW);
  __bf16* xn = (__bf16*)(ws + oXn);
  __bf16* xbT = (__bf16*)(ws + oXbT);
  __bf16* cn = (__bf16*)(ws + oCn);
  float* newc = (float*)(ws + oNewc);
  float* rowsum = (float*)(ws + oRowsum);
  float* counts = (float*)(ws + oCounts);
  int* keepInt = (int*)(ws + oKeep);
  int* shiftInt = (int*)(ws + oShift);
  int* done = (int*)(ws + oDone);

  k_init<<<dim3(NPTS), dim3(DIM), 0, stream>>>(x, cent, xn, xbT, shiftInt, done);

  for (int it = 0; it < MSITERS; ++it) {
    k_reset_iter<<<dim3(NPTS / 256), dim3(256), 0, stream>>>(rowsum, shiftInt);
    k_norm_c<<<dim3(NPTS), dim3(DIM), 0, stream>>>(cent, cn);
    k_gemm1<<<dim3(NPTS / 128, NPTS / 128), dim3(256), 0, stream>>>(cn, xn, Wm, rowsum, sigma);
    k_gemm2<<<dim3(NPTS / 64), dim3(256), 0, stream>>>(Wm, xbT, rowsum, cent, newc, shiftInt);
    k_update<<<dim3((NPTS * DIM) / 256), dim3(256), 0, stream>>>(cent, newc, shiftInt, done);
    k_done<<<dim3(1), dim3(1), 0, stream>>>(shiftInt, done);
  }

  // NMS on final centroids
  k_norm_c<<<dim3(NPTS), dim3(DIM), 0, stream>>>(cent, cn);
  k_reset_nms<<<dim3(NPTS / 256), dim3(256), 0, stream>>>(counts, keepInt);
  k_closest<<<dim3(NPTS / 128), dim3(256), 0, stream>>>(cn, xn, counts);
  k_map<<<dim3(NPTS / 128), dim3(256), 0, stream>>>(cn, counts, keepInt, sigma);
  k_keep<<<dim3(NPTS / 256), dim3(256), 0, stream>>>(keepInt, keepOut);
}
